// GatedDeltaNetDecode_37417755083576
// MI455X (gfx1250) — matrix operand order repaired and checked by reading, not compiled, not run on hardware
//
#include <hip/hip_runtime.h>
#include <hip/hip_fp16.h>

typedef __attribute__((ext_vector_type(16))) _Float16 v16h;
typedef __attribute__((ext_vector_type(8)))  float    v8f;

#define H_DIM     1536
#define HK        8
#define HV        16
#define DK        64
#define DV        64
#define KEY_DIM   512
#define VALUE_DIM 1024
#define CONV_DIM  2048

// workspace layout (float offsets)
#define WS_QKVZAB 0      // 3104 floats: qkv[2048] | z[1024] | a[16] | b[16]
#define WS_CONV   3104   // 2048 floats: conv_out
#define WS_GATED  5152   // 1024 floats: gated rmsnormed out
#define WS_QKDOT  6176   // 16
#define WS_EXPG   6192   // 16
#define WS_BETA   6208   // 16
#define WS_HALF   6224   // 2048 _Float16 = 1024 floats: qhat[16*64] | khat[16*64]

// ---------------------------------------------------------------------------
// K1: fused GEMV  [W_qkv; W_z; W_a; W_b] @ x  -> 3104 outputs
// One wave per row, x cached in LDS, float4 (global_load_b128) streaming.
// This kernel carries ~19 MB of the 25.7 MB total traffic.
// ---------------------------------------------------------------------------
__global__ __launch_bounds__(256) void k1_gemv(
    const float* __restrict__ x,
    const float* __restrict__ Wqkv, const float* __restrict__ Wz,
    const float* __restrict__ Wa,   const float* __restrict__ Wb,
    float* __restrict__ out) {
  __shared__ __align__(16) float xs[H_DIM];
  const int tid = threadIdx.x;
  for (int i = tid; i < H_DIM; i += 256) xs[i] = x[i];
  __syncthreads();

  const int lane = tid & 31, wave = tid >> 5;
  const int row = blockIdx.x * 8 + wave;          // 388 blocks * 8 = 3104 rows

  const float* Wrow;
  if (row < 2048)       Wrow = Wqkv + (size_t)row * H_DIM;
  else if (row < 3072)  Wrow = Wz   + (size_t)(row - 2048) * H_DIM;
  else if (row < 3088)  Wrow = Wa   + (size_t)(row - 3072) * H_DIM;
  else                  Wrow = Wb   + (size_t)(row - 3088) * H_DIM;

  const float4* W4 = (const float4*)Wrow;
  const float4* X4 = (const float4*)xs;
  float acc = 0.f;
#pragma unroll
  for (int i = 0; i < 12; i++) {                  // 12*32*4 = 1536 cols
    const int idx = lane + i * 32;
    const float4 w = W4[idx];
    const float4 xv = X4[idx];
    acc += w.x * xv.x + w.y * xv.y + w.z * xv.z + w.w * xv.w;
  }
  for (int o = 16; o > 0; o >>= 1) acc += __shfl_xor(acc, o, 32);
  if (lane == 0) out[row] = acc;
}

// ---------------------------------------------------------------------------
// K2: causal conv over cache(3 taps) + current token, then SiLU
// ---------------------------------------------------------------------------
__global__ __launch_bounds__(256) void k2_conv(
    const float* __restrict__ qkv, const float* __restrict__ cache,
    const float* __restrict__ cw, float* __restrict__ conv_out) {
  const int c = blockIdx.x * 256 + threadIdx.x;
  if (c >= CONV_DIM) return;
  float s = cache[c * 3 + 0] * cw[c * 4 + 0]
          + cache[c * 3 + 1] * cw[c * 4 + 1]
          + cache[c * 3 + 2] * cw[c * 4 + 2]
          + qkv[c]           * cw[c * 4 + 3];
  conv_out[c] = s / (1.f + __expf(-s));           // silu
}

// ---------------------------------------------------------------------------
// K3: per-Hk-head L2 normalization of q,k (q scaled by Dk^-0.5), f16 copies
// duplicated into the Hv layout, q.k dot; block 0 also computes beta and
// exp(g) scalars.
// ---------------------------------------------------------------------------
__global__ __launch_bounds__(64) void k3_prep(
    const float* __restrict__ conv_out,
    const float* __restrict__ abuf, const float* __restrict__ bbuf,
    const float* __restrict__ A_log, const float* __restrict__ dt_bias,
    _Float16* __restrict__ qhat, _Float16* __restrict__ khat,
    float* __restrict__ qkdot, float* __restrict__ expg,
    float* __restrict__ beta) {
  __shared__ float r1[64], r2[64];
  const int hk = blockIdx.x, t = threadIdx.x;
  const float qv = conv_out[hk * DK + t];
  const float kv = conv_out[KEY_DIM + hk * DK + t];
  r1[t] = qv * qv; r2[t] = kv * kv;
  __syncthreads();
  for (int s = 32; s > 0; s >>= 1) {
    if (t < s) { r1[t] += r1[t + s]; r2[t] += r2[t + s]; }
    __syncthreads();
  }
  const float qh = qv * rsqrtf(r1[0] + 1e-12f) * 0.125f;  // Dk^-0.5 = 1/8
  const float kh = kv * rsqrtf(r2[0] + 1e-12f);
  __syncthreads();
  r1[t] = qh * kh;
  __syncthreads();
  for (int s = 32; s > 0; s >>= 1) {
    if (t < s) r1[t] += r1[t + s];
    __syncthreads();
  }
  const float qkd = r1[0];
  const int h2 = hk * 2;                           // kv_group replication (rep=2)
  qhat[h2 * DK + t] = (_Float16)qh; qhat[(h2 + 1) * DK + t] = (_Float16)qh;
  khat[h2 * DK + t] = (_Float16)kh; khat[(h2 + 1) * DK + t] = (_Float16)kh;
  if (t == 0) { qkdot[h2] = qkd; qkdot[h2 + 1] = qkd; }

  if (blockIdx.x == 0 && t < HV) {
    const float bb = bbuf[t];
    beta[t] = 1.f / (1.f + __expf(-bb));
    const float xx = abuf[t] + dt_bias[t];
    const float sp = (xx > 20.f) ? xx : log1pf(__expf(xx)); // softplus
    expg[t] = __expf(-__expf(A_log[t]) * sp);               // exp(g)
  }
}

// ---------------------------------------------------------------------------
// K4: delta-rule state update via WMMA. One wave per Hv head.
//   R[:,0] = st_dec^T khat ; R[:,1] = st_dec^T qhat  (st_dec = state*exp(g))
//   delta  = (v - R[:,0]) * beta
//   out    = R[:,1] + (qhat.khat) * delta           // == new_state^T q
//   gated  = norm_w * out * rsqrt(mean(out^2)+1e-6) * silu(z)
// WMMA F32_16X16X32_F16: A = 16(v-tile) x 32(k) of st_dec (f16),
// B columns {0:khat, 1:qhat, rest 0}. A and B share a contiguous k order, k = 16*(lane/16) + e.
// ---------------------------------------------------------------------------
__global__ __launch_bounds__(32) void k4_state(
    const float* __restrict__ state, const float* __restrict__ conv_out,
    const float* __restrict__ zbuf,
    const _Float16* __restrict__ qhat, const _Float16* __restrict__ khat,
    const float* __restrict__ qkdot, const float* __restrict__ expg,
    const float* __restrict__ beta, const float* __restrict__ norm_w,
    float* __restrict__ gated) {
  __shared__ float R[2][64];
  const int h = blockIdx.x, lane = threadIdx.x;
  const float eg = expg[h];

  // --- B operand: 32x16 f16; lanes 0-15 hold K=0..15 (2 per VGPR),
  //     lanes 16-31 hold K=16..31. Column n = lane%16.
  const int n = lane & 15;
  const int bkb = (lane < 16) ? 0 : 16;
  v16h b_lo = {}, b_hi = {};
  const _Float16* kk = khat + h * DK;
  const _Float16* qq = qhat + h * DK;
  if (n == 0) {
#pragma unroll
    for (int e = 0; e < 16; e++) { b_lo[e] = kk[bkb + e]; b_hi[e] = kk[32 + bkb + e]; }
  } else if (n == 1) {
#pragma unroll
    for (int e = 0; e < 16; e++) { b_lo[e] = qq[bkb + e]; b_hi[e] = qq[32 + bkb + e]; }
  }

  // --- A operand: 16x32 f16; lane holds row M=lane%16.
  //     lanes 0-15: elements 0..15 -> K 0..15
  //     lanes 16-31: elements 0..15 -> K 16..31
  const int m = lane & 15;
  const int akb = (lane < 16) ? 0 : 16;
  const float* st = state + (size_t)h * DK * DV;   // [k][v], v-stride 1

#pragma unroll
  for (int t = 0; t < 4; t++) {                    // four 16-row v-tiles
    const int v = t * 16 + m;
    v16h a_lo, a_hi;
#pragma unroll
    for (int e = 0; e < 16; e++) {
      const int kidx = akb + e;
      a_lo[e] = (_Float16)(st[kidx * DV + v] * eg);          // K 0..31
      a_hi[e] = (_Float16)(st[(kidx + 32) * DV + v] * eg);   // K 32..63
    }
    v8f c = {};
    c = __builtin_amdgcn_wmma_f32_16x16x32_f16(false, a_lo, false, b_lo,
                                               (short)0, c, false, false);
    c = __builtin_amdgcn_wmma_f32_16x16x32_f16(false, a_hi, false, b_hi,
                                               (short)0, c, false, false);
    if (n < 2) {                                   // only columns 0,1 are live
      const int mb = (lane < 16) ? 0 : 8;          // C: VGPR j -> M = j (+8)
#pragma unroll
      for (int j = 0; j < 8; j++) R[n][t * 16 + mb + j] = c[j];
    }
  }
  __syncthreads();

  // --- epilogue: delta rule + gated RMSNorm; each lane owns 2 v's
  const float bet = beta[h], qkd = qkdot[h];
  float os[2], ssum = 0.f;
#pragma unroll
  for (int p = 0; p < 2; p++) {
    const int v = lane + p * 32;
    const float vv = conv_out[2 * KEY_DIM + h * DV + v];
    const float dlt = (vv - R[0][v]) * bet;
    const float o = R[1][v] + qkd * dlt;
    os[p] = o; ssum += o * o;
  }
  for (int o = 16; o > 0; o >>= 1) ssum += __shfl_xor(ssum, o, 32);
  const float rs = rsqrtf(ssum * (1.f / 64.f) + 1e-6f);
#pragma unroll
  for (int p = 0; p < 2; p++) {
    const int v = lane + p * 32;
    const float zz = zbuf[h * DV + v];
    const float sz = zz / (1.f + __expf(-zz));     // silu(z)
    gated[h * DV + v] = norm_w[v] * os[p] * rs * sz;
  }
}

// ---------------------------------------------------------------------------
// K5: y = W_out[1536x1024] @ gated
// ---------------------------------------------------------------------------
__global__ __launch_bounds__(256) void k5_out(
    const float* __restrict__ gated, const float* __restrict__ Wout,
    float* __restrict__ y) {
  __shared__ __align__(16) float gs[VALUE_DIM];
  const int tid = threadIdx.x;
  for (int i = tid; i < VALUE_DIM; i += 256) gs[i] = gated[i];
  __syncthreads();
  const int lane = tid & 31, wave = tid >> 5;
  const int row = blockIdx.x * 8 + wave;           // 192 blocks * 8 = 1536
  const float4* W4 = (const float4*)(Wout + (size_t)row * VALUE_DIM);
  const float4* G4 = (const float4*)gs;
  float acc = 0.f;
#pragma unroll
  for (int i = 0; i < 8; i++) {                    // 8*32*4 = 1024 cols
    const int idx = lane + i * 32;
    const float4 w = W4[idx];
    const float4 g = G4[idx];
    acc += w.x * g.x + w.y * g.y + w.z * g.z + w.w * g.w;
  }
  for (int o = 16; o > 0; o >>= 1) acc += __shfl_xor(acc, o, 32);
  if (lane == 0) y[row] = acc;
}

// ---------------------------------------------------------------------------
extern "C" void kernel_launch(void* const* d_in, const int* in_sizes, int n_in,
                              void* d_out, int out_size, void* d_ws, size_t ws_size,
                              hipStream_t stream) {
  const float* x     = (const float*)d_in[0];
  const float* Wqkv  = (const float*)d_in[1];
  const float* Wz    = (const float*)d_in[2];
  const float* Wa    = (const float*)d_in[3];
  const float* Wb    = (const float*)d_in[4];
  const float* Wout  = (const float*)d_in[5];
  const float* convw = (const float*)d_in[6];
  const float* A_log = (const float*)d_in[7];
  const float* dtb   = (const float*)d_in[8];
  const float* nw    = (const float*)d_in[9];
  const float* state = (const float*)d_in[10];
  const float* cache = (const float*)d_in[11];
  float* y  = (float*)d_out;
  float* ws = (float*)d_ws;

  float* qkvzab = ws + WS_QKVZAB;
  float* convo  = ws + WS_CONV;
  float* gated  = ws + WS_GATED;
  float* qkdot  = ws + WS_QKDOT;
  float* expg   = ws + WS_EXPG;
  float* betab  = ws + WS_BETA;
  _Float16* qhat = (_Float16*)(ws + WS_HALF);
  _Float16* khat = qhat + HV * DK;

  k1_gemv<<<388, 256, 0, stream>>>(x, Wqkv, Wz, Wa, Wb, qkvzab);
  k2_conv<<<8, 256, 0, stream>>>(qkvzab, cache, convw, convo);
  k3_prep<<<8, 64, 0, stream>>>(convo, qkvzab + 3072, qkvzab + 3088,
                                A_log, dtb, qhat, khat, qkdot, expg, betab);
  k4_state<<<16, 32, 0, stream>>>(state, convo, qkvzab + 2048, qhat, khat,
                                  qkdot, expg, betab, nw, gated);
  k5_out<<<192, 256, 0, stream>>>(gated, Wout, y);
}
